// QLoRABigNet_34454227649198
// MI455X (gfx1250) — matrix operand order repaired and checked by reading, not compiled, not run on hardware
//
#include <hip/hip_runtime.h>

typedef __attribute__((ext_vector_type(16))) __bf16 v16bf;
typedef __attribute__((ext_vector_type(8)))  __bf16 v8bf;
typedef __attribute__((ext_vector_type(4)))  __bf16 v4bf;
typedef __attribute__((ext_vector_type(8)))  float  v8f;

// pointee type for the async-to-LDS builtin (exactly as clang prints it)
typedef int v4i_vs __attribute__((vector_size(4 * sizeof(int))));
typedef __attribute__((address_space(1))) v4i_vs* gptr_v4i;
typedef __attribute__((address_space(3))) v4i_vs* lptr_v4i;

#define D       1024
#define BATCH   16384
#define LAYERS  18
#define RANK    32
#define GROUP   16
#define LN_EPS  1e-5f

// ---------------- GEMM tile config ----------------
#define BM 128
#define BN 128
#define BK 32
#define LDX 40   // halfs per LDS row (32 + 8 pad) -> 80B row stride, 16B aligned chunks
#define LDW 40

// CDNA5 async global->LDS path (ASYNCcnt). Guarded: falls back to reg staging.
#if defined(__HIP_DEVICE_COMPILE__) &&                                        \
    __has_builtin(__builtin_amdgcn_global_load_async_to_lds_b128) &&          \
    __has_builtin(__builtin_amdgcn_s_wait_asynccnt)
#define USE_ASYNC_W 1
#else
#define USE_ASYNC_W 0
#endif

__device__ __forceinline__ v16bf frag16(const __bf16* p0, const __bf16* p1) {
    v8bf a = *(const v8bf*)p0;
    v8bf b = *(const v8bf*)p1;
    return __builtin_shufflevector(a, b, 0,1,2,3,4,5,6,7,8,9,10,11,12,13,14,15);
}

// -----------------------------------------------------------------------------
// Kernel 1: dequantize 4-bit base weights + fold LoRA:  Weff = (q-8)*s + B@A
// grid: (D/64, D/16, LAYERS), block 256.  Output bf16 row-major [L, out, in].
// -----------------------------------------------------------------------------
__global__ __launch_bounds__(256)
void dequant_lora_fold(const int* __restrict__ wq, const float* __restrict__ scales,
                       const float* __restrict__ lora_a, const float* __restrict__ lora_b,
                       unsigned short* __restrict__ weff_raw) {
    __shared__ __align__(16) float As[RANK * 72];   // A[r][i_local], stride 72 (288B, 16B mult)
    __shared__ float Bs[16 * RANK];                 // B[o_local][r]
    __bf16* weff = (__bf16*)weff_raw;

    const int l  = blockIdx.z;
    const int i0 = blockIdx.x * 64;
    const int o0 = blockIdx.y * 16;
    const int tid = threadIdx.x;

    for (int j = tid; j < RANK * 64; j += 256) {
        int r = j >> 6, i = j & 63;
        As[r * 72 + i] = lora_a[((size_t)l * RANK + r) * D + i0 + i];
    }
    for (int j = tid; j < 16 * RANK; j += 256) {
        int oo = j >> 5, r = j & 31;
        Bs[oo * RANK + r] = lora_b[((size_t)l * D + o0 + oo) * RANK + r];
    }
    __syncthreads();

    const int oo = tid >> 4;          // 0..15
    const int ii = (tid & 15) * 4;    // 0..60, stays inside one quant group
    const int o  = o0 + oo;
    const int i  = i0 + ii;

    const int4 q4 = *(const int4*)&wq[((size_t)l * D + o) * D + i];
    const float sc = scales[((size_t)l * D + o) * (D / GROUP) + (i >> 4)];
    float w0 = (float)(q4.x - 8) * sc;
    float w1 = (float)(q4.y - 8) * sc;
    float w2 = (float)(q4.z - 8) * sc;
    float w3 = (float)(q4.w - 8) * sc;

#pragma unroll
    for (int r = 0; r < RANK; ++r) {
        const float b = Bs[oo * RANK + r];
        const float4 a = *(const float4*)&As[r * 72 + ii];
        w0 = fmaf(b, a.x, w0);
        w1 = fmaf(b, a.y, w1);
        w2 = fmaf(b, a.z, w2);
        w3 = fmaf(b, a.w, w3);
    }
    v4bf outp = { (__bf16)w0, (__bf16)w1, (__bf16)w2, (__bf16)w3 };
    *(v4bf*)&weff[((size_t)l * D + o) * D + i] = outp;
}

// -----------------------------------------------------------------------------
// Kernel 2: Y[M,1024] = relu?( X[M,1024] @ Weff[1024,1024]^T + bias (+ residual) )
// bf16 WMMA 16x16x32, 128x128 block tile, 8 waves, 64x32 wave tile,
// double-buffered LDS; W tile via GLOBAL_LOAD_ASYNC_TO_LDS when available.
// -----------------------------------------------------------------------------
__global__ __launch_bounds__(256, 2)
void qlora_gemm(const float* __restrict__ X, const unsigned short* __restrict__ W_raw,
                const float* __restrict__ bias, const float* __restrict__ residual,
                float* __restrict__ Y, int apply_relu) {
    __shared__ __align__(16) __bf16 Xs[2][BM * LDX];
    __shared__ __align__(16) __bf16 Ws[2][BN * LDW];
    const __bf16* __restrict__ Wb = (const __bf16*)W_raw;

    const int tid  = threadIdx.x;
    const int lane = tid & 31;
    const int wave = tid >> 5;     // 0..7
    const int wm   = wave & 1;     // wave row (M): 0..1, 64 rows each
    const int wn   = wave >> 1;    // wave col (N): 0..3, 32 cols each
    const int lrow = lane & 15;
    const int sel  = lane >> 4;

    const int col0 = blockIdx.x * BN;
    const int row0 = blockIdx.y * BM;

    float4 xr[4];

    // ---- X tile: fp32 global -> regs (cvt bf16 later) ----
    auto gloadX = [&](int kk) {
        const int k0 = kk * BK;
#pragma unroll
        for (int j = 0; j < 4; ++j) {
            int cid = tid + j * 256;            // 0..1023
            int r = cid >> 3, c = cid & 7;      // 128 rows x 8 float4
            xr[j] = *(const float4*)(X + (size_t)(row0 + r) * D + k0 + c * 4);
        }
    };
    auto sstoreX = [&](int buf) {
#pragma unroll
        for (int j = 0; j < 4; ++j) {
            int cid = tid + j * 256;
            int r = cid >> 3, c = cid & 7;
            v4bf p = { (__bf16)xr[j].x, (__bf16)xr[j].y, (__bf16)xr[j].z, (__bf16)xr[j].w };
            *(v4bf*)&Xs[buf][r * LDX + c * 4] = p;
        }
    };

#if USE_ASYNC_W
    // ---- W tile: bf16, no conversion needed -> async DMA straight into LDS ----
    auto gloadW = [&](int kk, int buf) {
        const int k0 = kk * BK;
#pragma unroll
        for (int j = 0; j < 2; ++j) {
            int cid = tid + j * 256;            // 0..511
            int r = cid >> 2, c = cid & 3;      // 128 rows x 4 b128 chunks
            const __bf16* g = Wb + (size_t)(col0 + r) * D + k0 + c * 8;
            __bf16* l = &Ws[buf][r * LDW + c * 8];
            __builtin_amdgcn_global_load_async_to_lds_b128(
                (gptr_v4i)(void*)g, (lptr_v4i)(void*)l, 0, 0);
        }
    };
#else
    v8bf wr[2];
    auto gloadWreg = [&](int kk) {
        const int k0 = kk * BK;
#pragma unroll
        for (int j = 0; j < 2; ++j) {
            int cid = tid + j * 256;
            int r = cid >> 2, c = cid & 3;
            wr[j] = *(const v8bf*)(Wb + (size_t)(col0 + r) * D + k0 + c * 8);
        }
    };
    auto sstoreWreg = [&](int buf) {
#pragma unroll
        for (int j = 0; j < 2; ++j) {
            int cid = tid + j * 256;
            int r = cid >> 2, c = cid & 3;
            *(v8bf*)&Ws[buf][r * LDW + c * 8] = wr[j];
        }
    };
#endif

    v8f acc[4][2] = {};

    auto compute = [&](int buf) {
        v16bf afr[4];
#pragma unroll
        for (int t = 0; t < 4; ++t) {
            // 16-bit A 16x32 layout: lane<16 -> K {0..7,16..23}; lane>=16 -> K {8..15,24..31}
            const __bf16* p = &Xs[buf][(wm * 64 + t * 16 + lrow) * LDX + sel * 8];
            afr[t] = frag16(p, p + 16);
        }
        v16bf bfr[2];
#pragma unroll
        for (int u = 0; u < 2; ++u) {
            // B 32x16 layout: lane n holds col n%16, K {sel*8..sel*8+7, 16+sel*8..16+sel*8+7} (matches A)
            const __bf16* p = &Ws[buf][(wn * 32 + u * 16 + lrow) * LDW + sel * 8];
            bfr[u] = frag16(p, p + 16);
        }
#pragma unroll
        for (int t = 0; t < 4; ++t)
#pragma unroll
            for (int u = 0; u < 2; ++u)
                acc[t][u] = __builtin_amdgcn_wmma_f32_16x16x32_bf16(
                    false, afr[t], false, bfr[u], (short)0, acc[t][u], false, false);
    };

    // ---- prologue ----
#if USE_ASYNC_W
    gloadX(0);
    gloadW(0, 0);
    sstoreX(0);
    __builtin_amdgcn_s_wait_asynccnt(0);
#else
    gloadX(0);
    gloadWreg(0);
    sstoreX(0);
    sstoreWreg(0);
#endif
    __syncthreads();

    // ---- main K loop, double buffered ----
    for (int kk = 0; kk < D / BK; ++kk) {
        const int cb = kk & 1;
#if USE_ASYNC_W
        if (kk < D / BK - 1) {
            gloadX(kk + 1);
            gloadW(kk + 1, cb ^ 1);   // DMA next W tile straight into spare buffer
        }
        compute(cb);                  // overlap DMA/global latency with WMMA
        if (kk < D / BK - 1) sstoreX(cb ^ 1);
        __builtin_amdgcn_s_wait_asynccnt(0);   // my async LDS writes done before barrier
#else
        if (kk < D / BK - 1) { gloadX(kk + 1); gloadWreg(kk + 1); }
        compute(cb);
        if (kk < D / BK - 1) { sstoreX(cb ^ 1); sstoreWreg(cb ^ 1); }
#endif
        __syncthreads();
    }

    // ---- epilogue: C/D layout -> m = v + sel*8, n = lane&15 ----
#pragma unroll
    for (int t = 0; t < 4; ++t) {
        const int gmb = row0 + wm * 64 + t * 16 + sel * 8;
#pragma unroll
        for (int u = 0; u < 2; ++u) {
            const int gn = col0 + wn * 32 + u * 16 + lrow;
            const float bv = bias[gn];
            v8f c = acc[t][u];
#pragma unroll
            for (int v = 0; v < 8; ++v) {
                const size_t idx = (size_t)(gmb + v) * D + gn;
                float val = c[v] + bv;
                if (residual) val += residual[idx];
                if (apply_relu) val = fmaxf(val, 0.0f);
                Y[idx] = val;
            }
        }
    }
}

// -----------------------------------------------------------------------------
// Kernel 3: in-place LayerNorm per row of 1024.  One block (256 thr) per row.
// -----------------------------------------------------------------------------
__global__ __launch_bounds__(256)
void layernorm_inplace(float* __restrict__ h, const float* __restrict__ gamma,
                       const float* __restrict__ beta) {
    __shared__ float rs[8], rss[8];
    __shared__ float s_mean, s_rstd;
    const int row = blockIdx.x;
    const int tid = threadIdx.x;
    float* p = h + (size_t)row * D + tid * 4;

    float4 v = *(const float4*)p;
    float s  = v.x + v.y + v.z + v.w;
    float ss = v.x * v.x + v.y * v.y + v.z * v.z + v.w * v.w;
#pragma unroll
    for (int off = 16; off > 0; off >>= 1) {
        s  += __shfl_down(s, off, 32);
        ss += __shfl_down(ss, off, 32);
    }
    if ((tid & 31) == 0) { rs[tid >> 5] = s; rss[tid >> 5] = ss; }
    __syncthreads();
    if (tid == 0) {
        float ts = 0.f, tss = 0.f;
#pragma unroll
        for (int w = 0; w < 8; ++w) { ts += rs[w]; tss += rss[w]; }
        const float mean = ts * (1.0f / D);
        const float var  = tss * (1.0f / D) - mean * mean;
        s_mean = mean;
        s_rstd = rsqrtf(var + LN_EPS);
    }
    __syncthreads();
    const float mean = s_mean, rstd = s_rstd;
    const float4 g = *(const float4*)(gamma + tid * 4);
    const float4 b = *(const float4*)(beta + tid * 4);
    float4 o;
    o.x = (v.x - mean) * rstd * g.x + b.x;
    o.y = (v.y - mean) * rstd * g.y + b.y;
    o.z = (v.z - mean) * rstd * g.z + b.z;
    o.w = (v.w - mean) * rstd * g.w + b.w;
    *(float4*)p = o;
}

// -----------------------------------------------------------------------------
extern "C" void kernel_launch(void* const* d_in, const int* in_sizes, int n_in,
                              void* d_out, int out_size, void* d_ws, size_t ws_size,
                              hipStream_t stream) {
    const float* x      = (const float*)d_in[0];
    const int*   wq     = (const int*)d_in[1];
    const float* scales = (const float*)d_in[2];
    const float* bias   = (const float*)d_in[3];
    const float* lora_a = (const float*)d_in[4];
    const float* lora_b = (const float*)d_in[5];
    const float* ln_g   = (const float*)d_in[6];
    const float* ln_b   = (const float*)d_in[7];
    float* out = (float*)d_out;

    // workspace layout: [ Weff bf16: 18*1024*1024 ][ act0 ][ act1 ][ act2 ] (fp32 acts)
    char* ws = (char*)d_ws;
    unsigned short* weff = (unsigned short*)ws;
    size_t woff = (size_t)LAYERS * D * D * sizeof(unsigned short);
    float* act[3];
    act[0] = (float*)(ws + woff);
    act[1] = act[0] + (size_t)BATCH * D;
    act[2] = act[1] + (size_t)BATCH * D;

    // 1) build all effective bf16 weights (dequant + LoRA fold), all 18 layers
    dequant_lora_fold<<<dim3(D / 64, D / 16, LAYERS), 256, 0, stream>>>(
        wq, scales, lora_a, lora_b, weff);

    const dim3 ggrid(D / BN, BATCH / BM);   // (8, 128)
    const size_t wstride = (size_t)D * D;

    const float* cur = x;   // current block input (also residual source)
    int c = 2;              // buffer rotation index
    for (int j = 0; j < 6; ++j) {
        const int i = 3 * j;
        float* h1 = act[(c + 1) % 3];
        float* h2 = act[(c + 2) % 3];
        float* h3 = (j == 5) ? out : act[(c + 1) % 3];   // h1 dead once h2 exists

        qlora_gemm<<<ggrid, 256, 0, stream>>>(cur, weff + (size_t)i * wstride,
                                              bias + (size_t)i * D, nullptr, h1, 1);
        qlora_gemm<<<ggrid, 256, 0, stream>>>(h1, weff + (size_t)(i + 1) * wstride,
                                              bias + (size_t)(i + 1) * D, nullptr, h2, 1);
        qlora_gemm<<<ggrid, 256, 0, stream>>>(h2, weff + (size_t)(i + 2) * wstride,
                                              bias + (size_t)(i + 2) * D, cur, h3, 0);
        if (j < 5) {
            layernorm_inplace<<<BATCH, 256, 0, stream>>>(h3, ln_g + (size_t)j * D,
                                                         ln_b + (size_t)j * D);
            cur = h3;
            c = (c + 1) % 3;
        }
    }
}
